// Bi_LSTM_Attention_89386859364659
// MI455X (gfx1250) — matrix operand order repaired and checked by reading, not compiled, not run on hardware
//
#include <hip/hip_runtime.h>
#include <hip/hip_bf16.h>

typedef __attribute__((ext_vector_type(16))) __bf16 v16bf;
typedef __attribute__((ext_vector_type(8)))  float  v8f;

// Problem constants
constexpr int BB = 64;     // batch
constexpr int TT = 256;    // time
constexpr int EE = 300;    // embedding
constexpr int EP = 320;    // embedding padded to mult of 32
constexpr int UU = 512;    // LSTM units
constexpr int G4 = 2048;   // 4*U gate width
constexpr int LL = 7;      // classes

// ---------------- WMMA helpers ----------------

__device__ inline v8f wmma_bf16(v16bf a, v16bf b, v8f c) {
  return __builtin_amdgcn_wmma_f32_16x16x32_bf16(false, a, false, b, (short)0, c,
                                                 false, false);
}

// A fragment (16x32 bf16). Caller passes p = base + row*lda + kk + (lane>=16 ? 8 : 0).
// lane<16: elems0..7=K0..7, elems8..15=K16..23 (row M=lane)
// lane>=16: elems0..7=K8..15, elems8..15=K24..31 (row M=lane-16)
__device__ inline v16bf ldfragA(const __hip_bfloat16* p) {
  union { uint4 q[2]; v16bf v; } u;
  u.q[0] = *reinterpret_cast<const uint4*>(p);
  u.q[1] = *reinterpret_cast<const uint4*>(p + 16);
  return u.v;
}

// B fragment (32x16 bf16, stored as W^T [N][K]).
// lane<16: N=lane, elems0..7=K0..7, elems8..15=K16..23; lane>=16: N=lane-16, elems0..7=K8..15, elems8..15=K24..31.
// Caller passes p = Wt + n*ldb + kk + (lane>=16 ? 8 : 0).
__device__ inline v16bf ldfragB(const __hip_bfloat16* p) {
  union { uint4 q[2]; v16bf v; } u;
  u.q[0] = *reinterpret_cast<const uint4*>(p);
  u.q[1] = *reinterpret_cast<const uint4*>(p + 16);
  return u.v;
}

// One K-segment of the 64x64-per-block GEMM. A: [M][K] row-major (lda=K),
// Wt: [Ncols][K] row-major (ldb=K). arow already includes lane&15.
__device__ inline void gemm_seg(const __hip_bfloat16* __restrict__ Ab,
                                const __hip_bfloat16* __restrict__ Wb,
                                int arow, int n0, int K, int lane,
                                v8f& acc0, v8f& acc1) {
  const int halfA = (lane >> 4) * 8;
  const int halfB = (lane >> 4) * 8;
  const int l16   = lane & 15;
  const __hip_bfloat16* ap  = Ab + (size_t)arow * K;
  const __hip_bfloat16* b0p = Wb + (size_t)(n0 + l16) * K;
  const __hip_bfloat16* b1p = Wb + (size_t)(n0 + 16 + l16) * K;
#pragma unroll 4
  for (int kk = 0; kk < K; kk += 32) {
    v16bf a  = ldfragA(ap  + kk + halfA);
    v16bf b0 = ldfragB(b0p + kk + halfB);
    v16bf b1 = ldfragB(b1p + kk + halfB);
    acc0 = wmma_bf16(a, b0, acc0);
    acc1 = wmma_bf16(a, b1, acc1);
  }
}

// ---------------- conversion / setup kernels ----------------

// k0 [2][300][2048] f32 -> k0t [2][2048][320] bf16, gate-permuted cols, K zero-padded
__global__ __launch_bounds__(256) void conv_k0(const float* __restrict__ k0,
                                               __hip_bfloat16* __restrict__ dst) {
  size_t idx = (size_t)blockIdx.x * 256 + threadIdx.x;
  const size_t total = 2ull * G4 * EP;
  if (idx >= total) return;
  int k = (int)(idx % EP);
  size_t r = idx / EP;
  int np = (int)(r % G4);
  int dir = (int)(r / G4);
  int nsrc = (np & 3) * UU + (np >> 2);
  float v = (k < EE) ? k0[((size_t)dir * EE + k) * G4 + nsrc] : 0.f;
  dst[idx] = __float2bfloat16(v);
}

// k12 [2][2][512][2048] -> k12t [layer][dir][2048][512] bf16 permuted
__global__ __launch_bounds__(256) void conv_k12(const float* __restrict__ k12,
                                                __hip_bfloat16* __restrict__ dst) {
  size_t idx = (size_t)blockIdx.x * 256 + threadIdx.x;
  const size_t total = 2ull * 2 * G4 * UU;
  if (idx >= total) return;
  int k = (int)(idx % UU);
  size_t r = idx / UU;
  int np = (int)(r % G4);
  size_t r2 = r / G4;
  int dir = (int)(r2 % 2);
  int layer = (int)(r2 / 2);
  int nsrc = (np & 3) * UU + (np >> 2);
  float v = k12[(((size_t)dir * 2 + layer) * UU + k) * G4 + nsrc];
  dst[idx] = __float2bfloat16(v);
}

// rec [2][3][512][2048] -> rect [layer][dir][2048][512] bf16 permuted
__global__ __launch_bounds__(256) void conv_rec(const float* __restrict__ rec,
                                                __hip_bfloat16* __restrict__ dst) {
  size_t idx = (size_t)blockIdx.x * 256 + threadIdx.x;
  const size_t total = 3ull * 2 * G4 * UU;
  if (idx >= total) return;
  int k = (int)(idx % UU);
  size_t r = idx / UU;
  int np = (int)(r % G4);
  size_t r2 = r / G4;
  int dir = (int)(r2 % 2);
  int layer = (int)(r2 / 2);
  int nsrc = (np & 3) * UU + (np >> 2);
  float v = rec[(((size_t)dir * 3 + layer) * UU + k) * G4 + nsrc];
  dst[idx] = __float2bfloat16(v);
}

// bias [2][3][2048] -> biasp [layer][dir][2048] f32 permuted
__global__ __launch_bounds__(256) void conv_bias(const float* __restrict__ bias,
                                                 float* __restrict__ dst) {
  size_t idx = (size_t)blockIdx.x * 256 + threadIdx.x;
  const size_t total = 3ull * 2 * G4;
  if (idx >= total) return;
  int np = (int)(idx % G4);
  size_t r = idx / G4;
  int dir = (int)(r % 2);
  int layer = (int)(r / 2);
  int nsrc = (np & 3) * UU + (np >> 2);
  dst[idx] = bias[((size_t)dir * 3 + layer) * G4 + nsrc];
}

// embedding gather -> x_bf16 [B*T][EP] (zero padded cols 300..319)
__global__ __launch_bounds__(256) void gather_x(const int* __restrict__ ids,
                                                const float* __restrict__ emb,
                                                __hip_bfloat16* __restrict__ dst) {
  size_t idx = (size_t)blockIdx.x * 256 + threadIdx.x;
  const size_t total = (size_t)BB * TT * EP;
  if (idx >= total) return;
  int e = (int)(idx % EP);
  size_t m = idx / EP;
  float v = 0.f;
  if (e < EE) v = emb[(size_t)ids[m] * EE + e];
  dst[idx] = __float2bfloat16(v);
}

__global__ __launch_bounds__(256) void zero_bf16(__hip_bfloat16* p, size_t n) {
  size_t i = (size_t)blockIdx.x * 256 + threadIdx.x;
  if (i < n) p[i] = __float2bfloat16(0.f);
}
__global__ __launch_bounds__(256) void zero_f32(float* p, size_t n) {
  size_t i = (size_t)blockIdx.x * 256 + threadIdx.x;
  if (i < n) p[i] = 0.f;
}

// ---------------- layer-0 input projection (big WMMA GEMM) ----------------
// xz0[dir][m][n'] = x[m] . k0t[dir][n'] + bias0[dir][n']   (m = b*T + t)
__global__ __launch_bounds__(256) void xz0_gemm(const __hip_bfloat16* __restrict__ X,
                                                const __hip_bfloat16* __restrict__ K0t,
                                                const float* __restrict__ bias0,
                                                float* __restrict__ xz0) {
  const int dir = blockIdx.z;
  const int mblk = blockIdx.y * 64;
  const int nblk = blockIdx.x * 64;
  const int tid = threadIdx.x;
  const int lane = tid & 31, wave = tid >> 5;
  const int half8 = (lane >> 4) * 8;
  const int l16 = lane & 15;
  const int m0 = (wave >> 1) * 16;
  const int nrel = (wave & 1) * 32;

  v8f acc0 = {}, acc1 = {};
  float b0 = bias0[(size_t)dir * G4 + nblk + nrel + l16];
  float b1 = bias0[(size_t)dir * G4 + nblk + nrel + 16 + l16];
#pragma unroll
  for (int r = 0; r < 8; r++) { acc0[r] = b0; acc1[r] = b1; }

  gemm_seg(X, K0t + (size_t)dir * G4 * EP,
           mblk + m0 + l16, nblk + nrel, EP, lane, acc0, acc1);

#pragma unroll
  for (int r = 0; r < 8; r++) {
    int m = mblk + m0 + r + half8;
    size_t row = ((size_t)dir * BB * TT + m) * G4;
    xz0[row + nblk + nrel + l16]      = acc0[r];
    xz0[row + nblk + nrel + 16 + l16] = acc1[r];
  }
}

// ---------------- fused LSTM step (WMMA GEMM + cell update) ----------------
// z = [acc init: xz0 slice OR bias] + (hIn @ Win) + (hSt @ Wrec); then gate math.
// Gate-permuted cols: n' = u*4 + {i,f,g,o} so one 64-col block owns 16 full units.
__global__ __launch_bounds__(256) void lstm_step(
    const __hip_bfloat16* __restrict__ hIn,   // [2][B][U] or null (layer 0)
    const __hip_bfloat16* __restrict__ Win,   // [2][G4][U] or null
    const __hip_bfloat16* __restrict__ hSt,   // [2][B][U]
    const __hip_bfloat16* __restrict__ Wrec,  // [2][G4][U]
    const float* __restrict__ xz0,            // [2][B*T][G4] or null
    const float* __restrict__ biasL,          // [2][G4] or null
    float* __restrict__ cst,                  // [2][B][U]
    __hip_bfloat16* __restrict__ hOut,        // [2][B][U] (ping-pong buffer)
    float* __restrict__ out3,                 // [2][B][T][U] or null (layer 2)
    int t) {
  const int dir = blockIdx.y;
  const int nblk = blockIdx.x * 64;
  const int tid = threadIdx.x;
  const int lane = tid & 31, wave = tid >> 5;
  const int half8 = (lane >> 4) * 8;
  const int l16 = lane & 15;
  const int m0 = (wave >> 1) * 16;
  const int nrel = (wave & 1) * 32;

  v8f acc0 = {}, acc1 = {};
  if (xz0) {
    const int tsrc = dir ? (TT - 1 - t) : t;
#pragma unroll
    for (int r = 0; r < 8; r++) {
      int b = m0 + r + half8;
      size_t row = ((size_t)dir * BB * TT + (size_t)b * TT + tsrc) * G4;
      acc0[r] = xz0[row + nblk + nrel + l16];
      acc1[r] = xz0[row + nblk + nrel + 16 + l16];
    }
  } else {
    float b0 = biasL[(size_t)dir * G4 + nblk + nrel + l16];
    float b1 = biasL[(size_t)dir * G4 + nblk + nrel + 16 + l16];
#pragma unroll
    for (int r = 0; r < 8; r++) { acc0[r] = b0; acc1[r] = b1; }
  }

  if (hIn)
    gemm_seg(hIn + (size_t)dir * BB * UU, Win + (size_t)dir * G4 * UU,
             m0 + l16, nblk + nrel, UU, lane, acc0, acc1);
  gemm_seg(hSt + (size_t)dir * BB * UU, Wrec + (size_t)dir * G4 * UU,
           m0 + l16, nblk + nrel, UU, lane, acc0, acc1);

  __shared__ float zsh[64][65];
#pragma unroll
  for (int r = 0; r < 8; r++) {
    int mrow = m0 + r + half8;
    zsh[mrow][nrel + l16]      = acc0[r];
    zsh[mrow][nrel + 16 + l16] = acc1[r];
  }
  __syncthreads();

  const int u0 = nblk >> 2;  // 16 units per block
#pragma unroll
  for (int j = 0; j < 4; j++) {
    int idx = tid + j * 256;
    int b = idx >> 4, ui = idx & 15;
    float zi = zsh[b][ui * 4 + 0];
    float zf = zsh[b][ui * 4 + 1];
    float zg = zsh[b][ui * 4 + 2];
    float zo = zsh[b][ui * 4 + 3];
    size_t si = ((size_t)dir * BB + b) * UU + u0 + ui;
    float c = cst[si];
    float ig = 1.f / (1.f + __expf(-zi));
    float fg = 1.f / (1.f + __expf(-zf));
    float og = 1.f / (1.f + __expf(-zo));
    float gs = zg / (1.f + fabsf(zg));
    c = fg * c + ig * gs;
    float h = og * (c / (1.f + fabsf(c)));
    cst[si] = c;
    hOut[si] = __float2bfloat16(h);
    if (out3) out3[(((size_t)dir * BB + b) * TT + t) * UU + u0 + ui] = h;
  }
}

// ---------------- attention + head ----------------

__device__ inline size_t o3idx(int dir, int b, int t, int u) {
  return (((size_t)dir * BB + b) * TT + t) * UU + u;
}

__global__ __launch_bounds__(256) void att_scores(const float* __restrict__ out3,
                                                  const float* __restrict__ Wa0,
                                                  const float* __restrict__ ba0,
                                                  const float* __restrict__ alpha_a,
                                                  const float* __restrict__ Wa1,
                                                  const float* __restrict__ ba1,
                                                  float* __restrict__ scores) {
  const int t = blockIdx.x, b = blockIdx.y, tid = threadIdx.x;
  __shared__ float es[UU];
  __shared__ float red[256];
  for (int u = tid; u < UU; u += 256)
    es[u] = 0.5f * (out3[o3idx(0, b, t, u)] + out3[o3idx(1, b, t, u)]);
  __syncthreads();
  float acc = ba0[tid];
  for (int k = 0; k < UU; k++) acc += es[k] * Wa0[(size_t)k * 256 + tid];
  float a = (acc >= 0.f) ? acc : alpha_a[tid] * acc;
  red[tid] = a * Wa1[tid];
  __syncthreads();
  for (int s = 128; s > 0; s >>= 1) {
    if (tid < s) red[tid] += red[tid + s];
    __syncthreads();
  }
  if (tid == 0) scores[(size_t)b * TT + t] = red[0] + ba1[0];
}

__global__ __launch_bounds__(256) void att_softmax(const float* __restrict__ scores,
                                                   float* __restrict__ w) {
  const int b = blockIdx.x, tid = threadIdx.x;
  __shared__ float red[256];
  float s = scores[(size_t)b * TT + tid];
  red[tid] = s;
  __syncthreads();
  for (int st = 128; st > 0; st >>= 1) {
    if (tid < st) red[tid] = fmaxf(red[tid], red[tid + st]);
    __syncthreads();
  }
  float mx = red[0];
  __syncthreads();
  float e = __expf(s - mx);
  red[tid] = e;
  __syncthreads();
  for (int st = 128; st > 0; st >>= 1) {
    if (tid < st) red[tid] += red[tid + st];
    __syncthreads();
  }
  w[(size_t)b * TT + tid] = e / red[0];
}

__global__ __launch_bounds__(256) void att_pool(const float* __restrict__ out3,
                                                const float* __restrict__ w,
                                                float* __restrict__ pooled) {
  const int b = blockIdx.x, tid = threadIdx.x;
  __shared__ float wsh[TT];
  wsh[tid] = w[(size_t)b * TT + tid];
  __syncthreads();
  for (int u = tid; u < UU; u += 256) {
    float acc = 0.f;
    for (int t = 0; t < TT; t++)
      acc += 0.5f * (out3[o3idx(0, b, t, u)] + out3[o3idx(1, b, t, u)]) * wsh[t];
    pooled[(size_t)b * UU + u] = acc;
  }
}

__global__ __launch_bounds__(256) void head_kernel(const float* __restrict__ pooled,
                                                   const float* __restrict__ Wd0,
                                                   const float* __restrict__ bd0,
                                                   const float* __restrict__ gamma,
                                                   const float* __restrict__ beta,
                                                   const float* __restrict__ mean,
                                                   const float* __restrict__ var,
                                                   const float* __restrict__ alpha_h,
                                                   const float* __restrict__ Wd1,
                                                   const float* __restrict__ bd1,
                                                   float* __restrict__ out) {
  const int b = blockIdx.x, tid = threadIdx.x;
  __shared__ float p[UU];
  __shared__ float hsh[256];
  __shared__ float lg[8];
  p[tid] = pooled[(size_t)b * UU + tid];
  p[tid + 256] = pooled[(size_t)b * UU + 256 + tid];
  __syncthreads();
  float acc = bd0[tid];
  for (int k = 0; k < UU; k++) acc += p[k] * Wd0[(size_t)k * 256 + tid];
  acc = (acc - mean[tid]) * rsqrtf(var[tid] + 1e-3f) * gamma[tid] + beta[tid];
  acc = (acc >= 0.f) ? acc : alpha_h[tid] * acc;
  hsh[tid] = acc;
  __syncthreads();
  if (tid < LL) {
    float l = bd1[tid];
    for (int k = 0; k < 256; k++) l += hsh[k] * Wd1[(size_t)k * LL + tid];
    lg[tid] = l;
  }
  __syncthreads();
  if (tid == 0) {
    float mx = lg[0];
    for (int j = 1; j < LL; j++) mx = fmaxf(mx, lg[j]);
    float e[LL], s = 0.f;
    for (int j = 0; j < LL; j++) { e[j] = __expf(lg[j] - mx); s += e[j]; }
    for (int j = 0; j < LL; j++) out[(size_t)b * LL + j] = e[j] / s;
  }
}

// ---------------- host launch ----------------

extern "C" void kernel_launch(void* const* d_in, const int* in_sizes, int n_in,
                              void* d_out, int out_size, void* d_ws, size_t ws_size,
                              hipStream_t stream) {
  const int*   ids     = (const int*)  d_in[0];
  const float* emb     = (const float*)d_in[1];
  const float* k0      = (const float*)d_in[2];
  const float* k12     = (const float*)d_in[3];
  const float* rec     = (const float*)d_in[4];
  const float* bias    = (const float*)d_in[5];
  const float* Wa0     = (const float*)d_in[6];
  const float* ba0     = (const float*)d_in[7];
  const float* alpha_a = (const float*)d_in[8];
  const float* Wa1     = (const float*)d_in[9];
  const float* ba1     = (const float*)d_in[10];
  const float* Wd0     = (const float*)d_in[11];
  const float* bd0     = (const float*)d_in[12];
  const float* gamma   = (const float*)d_in[13];
  const float* beta    = (const float*)d_in[14];
  const float* bn_mean = (const float*)d_in[15];
  const float* bn_var  = (const float*)d_in[16];
  const float* alpha_h = (const float*)d_in[17];
  const float* Wd1     = (const float*)d_in[18];
  const float* bd1     = (const float*)d_in[19];
  float* out = (float*)d_out;

  // workspace layout (all sizes multiples of 256B)
  char* base = (char*)d_ws;
  size_t off = 0;
  auto take = [&](size_t sz) { void* p = base + off; off += sz; return p; };
  __hip_bfloat16* k0t   = (__hip_bfloat16*)take(2ull * G4 * EP * 2);
  __hip_bfloat16* k12t  = (__hip_bfloat16*)take(2ull * 2 * G4 * UU * 2);
  __hip_bfloat16* rect  = (__hip_bfloat16*)take(3ull * 2 * G4 * UU * 2);
  float*          biasp = (float*)take(3ull * 2 * G4 * 4);
  __hip_bfloat16* xb    = (__hip_bfloat16*)take((size_t)BB * TT * EP * 2);
  float*          xz0   = (float*)take(2ull * BB * TT * G4 * 4);
  __hip_bfloat16* hbuf  = (__hip_bfloat16*)take(3ull * 2 * 2 * BB * UU * 2);
  float*          cst   = (float*)take(3ull * 2 * BB * UU * 4);
  float*          o3    = (float*)take(2ull * BB * TT * UU * 4);
  float*          sc    = (float*)take((size_t)BB * TT * 4);
  float*          wsm   = (float*)take((size_t)BB * TT * 4);
  float*          pool  = (float*)take((size_t)BB * UU * 4);
  (void)ws_size; (void)n_in; (void)in_sizes; (void)out_size;

  auto grid1 = [](size_t n) { return dim3((unsigned)((n + 255) / 256)); };

  // weight conversion (bf16, [N][K] transposed, gate-permuted)
  conv_k0 <<<grid1(2ull * G4 * EP),      256, 0, stream>>>(k0,  k0t);
  conv_k12<<<grid1(2ull * 2 * G4 * UU),  256, 0, stream>>>(k12, k12t);
  conv_rec<<<grid1(3ull * 2 * G4 * UU),  256, 0, stream>>>(rec, rect);
  conv_bias<<<grid1(3ull * 2 * G4),      256, 0, stream>>>(bias, biasp);
  gather_x<<<grid1((size_t)BB * TT * EP), 256, 0, stream>>>(ids, emb, xb);
  zero_bf16<<<grid1(3ull * 2 * 2 * BB * UU), 256, 0, stream>>>(hbuf, 3ull * 2 * 2 * BB * UU);
  zero_f32 <<<grid1(3ull * 2 * BB * UU),     256, 0, stream>>>(cst, 3ull * 2 * BB * UU);

  // layer-0 input projection: xz0 = x @ k0 + bias0  (WMMA, fills the GPU)
  xz0_gemm<<<dim3(G4 / 64, (BB * TT) / 64, 2), 256, 0, stream>>>(xb, k0t, biasp, xz0);

  // recurrent scan: 3 fused WMMA+cell kernels per step, both dirs per launch
  const size_t hstride = 2ull * BB * UU;       // per (layer,parity) slab
  const size_t wstride = 2ull * G4 * UU;       // per layer slab of rect/k12t
  const size_t bstride = 2ull * G4;            // per layer slab of biasp
  for (int t = 0; t < TT; t++) {
    int p = t & 1;
    __hip_bfloat16* h0r = hbuf + (0 * 2 + p) * hstride;
    __hip_bfloat16* h0w = hbuf + (0 * 2 + (1 - p)) * hstride;
    __hip_bfloat16* h1r = hbuf + (1 * 2 + p) * hstride;
    __hip_bfloat16* h1w = hbuf + (1 * 2 + (1 - p)) * hstride;
    __hip_bfloat16* h2r = hbuf + (2 * 2 + p) * hstride;
    __hip_bfloat16* h2w = hbuf + (2 * 2 + (1 - p)) * hstride;

    lstm_step<<<dim3(G4 / 64, 2), 256, 0, stream>>>(
        nullptr, nullptr, h0r, rect + 0 * wstride, xz0, nullptr,
        cst + 0 * hstride, h0w, nullptr, t);
    lstm_step<<<dim3(G4 / 64, 2), 256, 0, stream>>>(
        h0w, k12t + 0 * wstride, h1r, rect + 1 * wstride, nullptr,
        biasp + 1 * bstride, cst + 1 * hstride, h1w, nullptr, t);
    lstm_step<<<dim3(G4 / 64, 2), 256, 0, stream>>>(
        h1w, k12t + 1 * wstride, h2r, rect + 2 * wstride, nullptr,
        biasp + 2 * bstride, cst + 2 * hstride, h2w, o3, t);
  }

  // attention pooling + classification head
  att_scores<<<dim3(TT, BB), 256, 0, stream>>>(o3, Wa0, ba0, alpha_a, Wa1, ba1, sc);
  att_softmax<<<BB, 256, 0, stream>>>(sc, wsm);
  att_pool<<<BB, 256, 0, stream>>>(o3, wsm, pool);
  head_kernel<<<BB, 256, 0, stream>>>(pool, Wd0, bd0, gamma, beta, bn_mean, bn_var,
                                      alpha_h, Wd1, bd1, out);
}
